// MultiHeadSelfAttention_38062000177387
// MI455X (gfx1250) — matrix operand order repaired and checked by reading, not compiled, not run on hardware
//
#include <hip/hip_runtime.h>

typedef __attribute__((ext_vector_type(16))) _Float16 v16h;
typedef __attribute__((ext_vector_type(8)))  _Float16 v8h;
typedef __attribute__((ext_vector_type(4)))  _Float16 v4h;
typedef __attribute__((ext_vector_type(2)))  _Float16 v2h;
typedef __attribute__((ext_vector_type(8)))  float    v8f;
typedef __attribute__((ext_vector_type(4)))  float    v4f;

static constexpr int Bn   = 4;
static constexpr int Tn   = 2048;
static constexpr int Dn   = 512;
static constexpr int Hn   = 8;
static constexpr int HDn  = 64;
static constexpr int Mtot = Bn * Tn;   // 8192 rows

// Assemble a v16h A/B fragment from two contiguous 8-half (16B) chunks.
__device__ __forceinline__ v16h make_frag(v8h lo, v8h hi) {
  v16h r;
#pragma unroll
  for (int i = 0; i < 8; ++i) { r[i] = lo[i]; r[i + 8] = hi[i]; }
  return r;
}

__device__ __forceinline__ v8f wmma_f16(v16h a, v16h b, v8f c) {
  return __builtin_amdgcn_wmma_f32_16x16x32_f16(false, a, false, b,
                                                (short)0, c, false, false);
}

// 4 elements per thread: global_load_b128 -> global_store_b64
__global__ void cvt_f32_to_f16_x4(const float* __restrict__ in,
                                  _Float16* __restrict__ out, int n4) {
  int i = blockIdx.x * blockDim.x + threadIdx.x;
  if (i < n4) {
    v4f f = ((const v4f*)in)[i];
    v4h h;
#pragma unroll
    for (int j = 0; j < 4; ++j) h[j] = (_Float16)f[j];
    ((v4h*)out)[i] = h;
  }
}

// Y = A @ W^T + bias, A:[M,K] f16 row-major, W:[512,K] f16 row-major.
// Per wave: 32(M) x 64(N) tile, K in steps of 32, software-pipelined.
// MODE 0: f16 -> [B,H,T,HD] (Q,K)
// MODE 1: f16 -> [B,H,HD,T] (V, transposed; T permuted within 32-blocks to
//         match the attention kernel's interleaved P-column packing:
//         physical t' = 2*(t&15) + ((t>>4)&1) within each block of 32)
// MODE 2: f32 -> [M,512] (output projection)
template <int MODE>
__global__ __launch_bounds__(256) void gemm_bias_wmma(
    const _Float16* __restrict__ A,
    const _Float16* __restrict__ W,
    const float* __restrict__ bias,
    void* __restrict__ out, int M, int K)
{
  const int lane = threadIdx.x & 31;
  const int wave = threadIdx.x >> 5;
  const int wid  = blockIdx.x * 8 + wave;
  const int mtiles = M >> 5;                 // 32-row tiles
  const int mt = wid % mtiles;
  const int nt = wid / mtiles;
  const int mbase = mt << 5;
  const int nbase = nt << 6;
  const int lm = lane & 15, lh = lane >> 4;

  const _Float16* arow0 = A + (size_t)(mbase + lm) * K;
  const _Float16* arow1 = A + (size_t)(mbase + 16 + lm) * K;
  const _Float16* wrow[4];
#pragma unroll
  for (int t = 0; t < 4; ++t)
    wrow[t] = W + (size_t)(nbase + t * 16 + lm) * K + lh * 16;
  const int ka = lh * 16;

  v8f acc[2][4] = {};

  // prologue: fragments for kb = 0
  v16h af0 = make_frag(*(const v8h*)(arow0 + ka), *(const v8h*)(arow0 + 8 + ka));
  v16h af1 = make_frag(*(const v8h*)(arow1 + ka), *(const v8h*)(arow1 + 8 + ka));
  v16h bf0 = *(const v16h*)(wrow[0]);
  v16h bf1 = *(const v16h*)(wrow[1]);
  v16h bf2 = *(const v16h*)(wrow[2]);
  v16h bf3 = *(const v16h*)(wrow[3]);

#pragma unroll 2
  for (int kb = 0; kb < K; kb += 32) {
    // issue next k-step loads into fresh registers (no wait yet)
    v16h naf0, naf1, nbf0, nbf1, nbf2, nbf3;
    const int kn = kb + 32;
    if (kn < K) {
      naf0 = make_frag(*(const v8h*)(arow0 + kn + ka),
                       *(const v8h*)(arow0 + kn + 8 + ka));
      naf1 = make_frag(*(const v8h*)(arow1 + kn + ka),
                       *(const v8h*)(arow1 + kn + 8 + ka));
      nbf0 = *(const v16h*)(wrow[0] + kn);
      nbf1 = *(const v16h*)(wrow[1] + kn);
      nbf2 = *(const v16h*)(wrow[2] + kn);
      nbf3 = *(const v16h*)(wrow[3] + kn);
      if (kn + 32 < K) __builtin_prefetch(arow0 + kn + 32, 0, 3);
    }
    // 8 WMMAs on current fragments (overlap with the loads above)
    acc[0][0] = wmma_f16(af0, bf0, acc[0][0]);
    acc[0][1] = wmma_f16(af0, bf1, acc[0][1]);
    acc[0][2] = wmma_f16(af0, bf2, acc[0][2]);
    acc[0][3] = wmma_f16(af0, bf3, acc[0][3]);
    acc[1][0] = wmma_f16(af1, bf0, acc[1][0]);
    acc[1][1] = wmma_f16(af1, bf1, acc[1][1]);
    acc[1][2] = wmma_f16(af1, bf2, acc[1][2]);
    acc[1][3] = wmma_f16(af1, bf3, acc[1][3]);
    if (kn < K) {
      af0 = naf0; af1 = naf1;
      bf0 = nbf0; bf1 = nbf1; bf2 = nbf2; bf3 = nbf3;
    }
  }

#pragma unroll
  for (int t = 0; t < 4; ++t) {
    const int n  = nbase + t * 16 + lm;
    const float bv = bias[n];
#pragma unroll
    for (int g = 0; g < 2; ++g) {
#pragma unroll
      for (int r = 0; r < 8; ++r) {
        const int m = mbase + g * 16 + r + 8 * lh;  // C/D layout row
        const float v = acc[g][t][r] + bv;
        if (MODE == 2) {
          ((float*)out)[(size_t)m * Dn + n] = v;
        } else {
          const int b  = m >> 11;               // T = 2048
          const int tq = m & (Tn - 1);
          const int h  = n >> 6;                // HD = 64
          const int hd = n & 63;
          _Float16* o = (_Float16*)out;
          if (MODE == 0) {
            o[(((size_t)b * Hn + h) * Tn + tq) * HDn + hd] = (_Float16)v;
          } else {
            // interleaved key permutation within each 32-block (see attention)
            const int tq_p = (tq & ~31) | (((tq & 15) << 1) | ((tq >> 4) & 1));
            o[(((size_t)b * Hn + h) * HDn + hd) * Tn + tq_p] = (_Float16)v;
          }
        }
      }
    }
  }
}

// Flash attention: one wave per 16-query tile, online softmax over 32-key blocks.
// Q,K: [B,H,T,HD] f16; Vt: [B,H,HD,T] f16 (T interleave-permuted per 32-block);
// O: [B*T, D] f16.
// - Q prescaled by 0.125*log2(e): softmax computed base-2 with raw v_exp_f32.
// - Row sums come from a 5th WMMA against an all-ones B fragment (no shuffle tree).
// - P packed as (col 2*lm, 2*lm+1) pairs -> 8x ds_store_b32; V rows permuted to match.
__global__ __launch_bounds__(256) void attn_flash_wmma(
    const _Float16* __restrict__ Q,
    const _Float16* __restrict__ Km,
    const _Float16* __restrict__ Vt,
    _Float16* __restrict__ O)
{
  __shared__ __align__(32) _Float16 Pl[8][16 * 32];   // per-wave P transpose buffer

  const int lane = threadIdx.x & 31;
  const int wave = threadIdx.x >> 5;
  const int tid  = blockIdx.x * 8 + wave;  // 0..4095
  const int qt   = tid & 127;              // query tile within (b,h)
  const int bh   = tid >> 7;               // 0..31
  const int h = bh & 7, b = bh >> 3;
  const int lm = lane & 15, lh = lane >> 4;

  const _Float16* Qp = Q  + ((size_t)bh * Tn + qt * 16) * HDn;
  const _Float16* Kp = Km + (size_t)bh * Tn * HDn;
  const _Float16* Vp = Vt + (size_t)bh * HDn * Tn;

  // Q A-fragments (16 queries x 64 hd -> two k=32 fragments), loaded once,
  // prescaled by 1/sqrt(HD) * log2(e) so scores are directly exp2 arguments.
  const _Float16 qs = (_Float16)(0.125f * 1.44269504088896f);
  const _Float16* qrow = Qp + lm * HDn;
  v16h qa0 = make_frag(*(const v8h*)(qrow + lh * 16),
                       *(const v8h*)(qrow + 8 + lh * 16));
  v16h qa1 = make_frag(*(const v8h*)(qrow + 32 + lh * 16),
                       *(const v8h*)(qrow + 40 + lh * 16));
#pragma unroll
  for (int i = 0; i < 16; ++i) { qa0[i] *= qs; qa1[i] *= qs; }

  v16h ones;
#pragma unroll
  for (int i = 0; i < 16; ++i) ones[i] = (_Float16)1.0f;

  // Per-lane K row bases: kf(coltile c, kc) <- Kp[(kb + c*16 + lm)*HD + kc + lh*16]
  const _Float16* krow0 = Kp + (size_t)lm * HDn + lh * 16;        // coltile 0
  const _Float16* krow1 = Kp + (size_t)(16 + lm) * HDn + lh * 16; // coltile 1
  const _Float16* vrow[4];
#pragma unroll
  for (int t = 0; t < 4; ++t)
    vrow[t] = Vp + (size_t)(t * 16 + lm) * Tn + lh * 16;

  float mrow[8];
#pragma unroll
  for (int r = 0; r < 8; ++r) mrow[r] = -1.0e30f;
  v8f oacc[4] = {};
  v8f lacc = {};                 // row sums of P, accumulated by WMMA vs ones

  _Float16* P = Pl[wave];

  // prologue: K fragments for kb = 0
  v16h kf0 = *(const v16h*)(krow0);
  v16h kf1 = *(const v16h*)(krow0 + 32);
  v16h kf2 = *(const v16h*)(krow1);
  v16h kf3 = *(const v16h*)(krow1 + 32);

#pragma unroll 2
  for (int kb = 0; kb < Tn; kb += 32) {
    // V fragments for this block: independent of everything until the last WMMAs
    v16h vf0 = *(const v16h*)(vrow[0] + kb);
    v16h vf1 = *(const v16h*)(vrow[1] + kb);
    v16h vf2 = *(const v16h*)(vrow[2] + kb);
    v16h vf3 = *(const v16h*)(vrow[3] + kb);

    // S = Q K^T (already in log2 units): 16 queries x 32 keys
    v8f s0 = {}, s1 = {};
    s0 = wmma_f16(qa0, kf0, s0);
    s0 = wmma_f16(qa1, kf1, s0);
    s1 = wmma_f16(qa0, kf2, s1);
    s1 = wmma_f16(qa1, kf3, s1);

    // issue next block's K fragments now; they complete under the softmax VALU
    const int kn = kb + 32;
    v16h nk0, nk1, nk2, nk3;
    if (kn < Tn) {
      const size_t koff = (size_t)kn * HDn;
      nk0 = *(const v16h*)(krow0 + koff);
      nk1 = *(const v16h*)(krow0 + koff + 32);
      nk2 = *(const v16h*)(krow1 + koff);
      nk3 = *(const v16h*)(krow1 + koff + 32);
      __builtin_prefetch(vrow[0] + kn, 0, 3);
    }

    // ---- online softmax, base 2 (rows r+8*lh live across the half-wave lanes)
    float cmax[8];
#pragma unroll
    for (int r = 0; r < 8; ++r) cmax[r] = fmaxf(s0[r], s1[r]);
#pragma unroll
    for (int off = 1; off < 16; off <<= 1)
#pragma unroll
      for (int r = 0; r < 8; ++r)
        cmax[r] = fmaxf(cmax[r], __shfl_xor(cmax[r], off, 32));

    float alpha[8];
#pragma unroll
    for (int r = 0; r < 8; ++r) {
      const float mo = mrow[r];
      const float mn = fmaxf(mo, cmax[r]);
      mrow[r]  = mn;
      alpha[r] = __builtin_amdgcn_exp2f(mo - mn);
      s0[r] = __builtin_amdgcn_exp2f(s0[r] - mn);
      s1[r] = __builtin_amdgcn_exp2f(s1[r] - mn);
    }
#pragma unroll
    for (int t = 0; t < 4; ++t)
#pragma unroll
      for (int r = 0; r < 8; ++r) oacc[t][r] *= alpha[r];
#pragma unroll
    for (int r = 0; r < 8; ++r) lacc[r] *= alpha[r];

    // ---- P: pack col pair (2*lm, 2*lm+1) per row -> 8x ds_store_b32.
    // Physical col p maps to key (p>>1) + 16*(p&1); V rows are stored in the
    // same permuted order, so the P@V contraction is unchanged.
#pragma unroll
    for (int r = 0; r < 8; ++r) {
      const int row = r + 8 * lh;
      v2h pk;
      pk[0] = (_Float16)s0[r];
      pk[1] = (_Float16)s1[r];
      *(v2h*)(P + row * 32 + 2 * lm) = pk;
    }
    const v16h pf = make_frag(*(const v8h*)(P + lm * 32 + lh * 16),
                              *(const v8h*)(P + lm * 32 + 8 + lh * 16));

    // ---- O += P @ V ; row sums += P @ ones (5 WMMAs)
    oacc[0] = wmma_f16(pf, vf0, oacc[0]);
    oacc[1] = wmma_f16(pf, vf1, oacc[1]);
    oacc[2] = wmma_f16(pf, vf2, oacc[2]);
    oacc[3] = wmma_f16(pf, vf3, oacc[3]);
    lacc    = wmma_f16(pf, ones, lacc);

    if (kn < Tn) { kf0 = nk0; kf1 = nk1; kf2 = nk2; kf3 = nk3; }
  }

  // ---- epilogue: O /= rowsum, store f16 [B*T, D]
  float inv[8];
#pragma unroll
  for (int r = 0; r < 8; ++r) inv[r] = __builtin_amdgcn_rcpf(lacc[r]);
#pragma unroll
  for (int t = 0; t < 4; ++t) {
#pragma unroll
    for (int r = 0; r < 8; ++r) {
      const int m = b * Tn + qt * 16 + r + 8 * lh;
      const int n = h * HDn + t * 16 + lm;
      O[(size_t)m * Dn + n] = (_Float16)(oacc[t][r] * inv[r]);
    }
  }
}

extern "C" void kernel_launch(void* const* d_in, const int* in_sizes, int n_in,
                              void* d_out, int out_size, void* d_ws, size_t ws_size,
                              hipStream_t stream) {
  const float* x  = (const float*)d_in[0];
  const float* wq = (const float*)d_in[1];
  const float* bq = (const float*)d_in[2];
  const float* wk = (const float*)d_in[3];
  const float* bk = (const float*)d_in[4];
  const float* wv = (const float*)d_in[5];
  const float* bv = (const float*)d_in[6];
  const float* wo = (const float*)d_in[7];
  const float* bo = (const float*)d_in[8];

  // Workspace layout (element counts in f16 halves)
  const size_t nX = (size_t)Mtot * Dn;     // 8192*512
  const size_t nW = (size_t)Dn * Dn;       // 512*512
  _Float16* xh  = (_Float16*)d_ws;
  _Float16* wqh = xh  + nX;
  _Float16* wkh = wqh + nW;
  _Float16* wvh = wkh + nW;
  _Float16* woh = wvh + nW;
  _Float16* Qh  = woh + nW;                // [B,H,T,HD]
  _Float16* Kh  = Qh  + nX;                // [B,H,T,HD]
  _Float16* Vth = Kh  + nX;                // [B,H,HD,T] (T permuted per 32-block)
  _Float16* Oh  = Vth + nX;                // [B*T, D]

  // fp32 -> f16 down-conversion (x4 vectorized)
  cvt_f32_to_f16_x4<<<(int)(nX / 4 / 256), 256, 0, stream>>>(x,  xh,  (int)(nX / 4));
  cvt_f32_to_f16_x4<<<(int)(nW / 4 / 256), 256, 0, stream>>>(wq, wqh, (int)(nW / 4));
  cvt_f32_to_f16_x4<<<(int)(nW / 4 / 256), 256, 0, stream>>>(wk, wkh, (int)(nW / 4));
  cvt_f32_to_f16_x4<<<(int)(nW / 4 / 256), 256, 0, stream>>>(wv, wvh, (int)(nW / 4));
  cvt_f32_to_f16_x4<<<(int)(nW / 4 / 256), 256, 0, stream>>>(wo, woh, (int)(nW / 4));

  // Projections: (M/32 m-tiles)*(8 n-tiles) = 2048 waves -> 256 blocks of 8 waves
  gemm_bias_wmma<0><<<256, 256, 0, stream>>>(xh, wqh, bq, (void*)Qh,  Mtot, Dn);
  gemm_bias_wmma<0><<<256, 256, 0, stream>>>(xh, wkh, bk, (void*)Kh,  Mtot, Dn);
  gemm_bias_wmma<1><<<256, 256, 0, stream>>>(xh, wvh, bv, (void*)Vth, Mtot, Dn);

  // Flash attention: B*H*(T/16) = 4096 query tiles, one wave each
  attn_flash_wmma<<<512, 256, 0, stream>>>(Qh, Kh, Vth, Oh);

  // Output projection -> fp32
  gemm_bias_wmma<2><<<256, 256, 0, stream>>>(Oh, woh, bo, d_out, Mtot, Dn);
}
